// Attention_68221260529678
// MI455X (gfx1250) — matrix operand order repaired and checked by reading, not compiled, not run on hardware
//
#include <hip/hip_runtime.h>

typedef unsigned short u16;
typedef unsigned int   u32;

typedef __attribute__((ext_vector_type(16))) __bf16 v16bf;
typedef __attribute__((ext_vector_type(8)))  float  v8f;

// ---- problem constants (fixed by setup_inputs) ----
#define BB   2
#define TT   2048      // T == S
#define DD   2048
#define NQ   8
#define HH   256
#define NH   2048      // N*H
#define BT   4096      // B*T

union FragU { v16bf v; uint4 q[2]; u32 w[8]; };

__device__ __forceinline__ u16 f2bf(float x) {
    u32 u = __float_as_uint(x);
    u32 r = u + 0x7FFFu + ((u >> 16) & 1u);   // RNE
    return (u16)(r >> 16);
}
__device__ __forceinline__ float bf2f(u16 v) {
    return __uint_as_float(((u32)v) << 16);
}
__device__ __forceinline__ u32 pack2(float a, float b) {
    return (u32)f2bf(a) | ((u32)f2bf(b) << 16);
}
__device__ __forceinline__ v16bf ld_split(const u16* p0, const u16* p1) {
    FragU f;
    f.q[0] = *(const uint4*)p0;
    f.q[1] = *(const uint4*)p1;
    return f.v;
}
__device__ __forceinline__ v16bf ld_contig(const u16* p) {
    FragU f;
    const uint4* q = (const uint4*)p;
    f.q[0] = q[0];
    f.q[1] = q[1];
    return f.v;
}
__device__ __forceinline__ v8f wmma_bf16(v16bf a, v16bf b, v8f c) {
    return __builtin_amdgcn_wmma_f32_16x16x32_bf16(false, a, false, b, (short)0, c, false, false);
}

// ------------------------------------------------------------------
// 1) prep: x -> bf16 ; weights -> bf16 transposed [col][D] layouts
// ------------------------------------------------------------------
#define XCNT  (BB*TT*DD)          // 8388608
#define QWCNT (NH*DD)             // 4194304
#define KWCNT (HH*DD)             // 524288
#define OWCNT (DD*NH)             // 4194304
#define PREP_TOTAL (XCNT + QWCNT + 2*KWCNT + OWCNT)

__global__ void __launch_bounds__(256) prep_kernel(
    const float* __restrict__ x, const float* __restrict__ q_w,
    const float* __restrict__ kv_w, const float* __restrict__ out_w,
    u16* __restrict__ xbuf, u16* __restrict__ qwt, u16* __restrict__ kwt,
    u16* __restrict__ vwt, u16* __restrict__ owt) {
    long long idx = (long long)blockIdx.x * blockDim.x + threadIdx.x;
    if (idx >= PREP_TOTAL) return;
    if (idx < XCNT) { xbuf[idx] = f2bf(x[idx]); return; }
    idx -= XCNT;
    if (idx < QWCNT) {              // qwt[c][d] = q_w[n][d][h], c = n*H+h
        int c = (int)(idx / DD), d = (int)(idx % DD);
        qwt[idx] = f2bf(q_w[(size_t)(c >> 8) * DD * HH + (size_t)d * HH + (c & 255)]);
        return;
    }
    idx -= QWCNT;
    if (idx < KWCNT) {              // kwt[h][d] = kv_w[0][0][d][h]
        int h = (int)(idx / DD), d = (int)(idx % DD);
        kwt[idx] = f2bf(kv_w[(size_t)d * HH + h]);
        return;
    }
    idx -= KWCNT;
    if (idx < KWCNT) {              // vwt[h][d] = kv_w[1][0][d][h]
        int h = (int)(idx / DD), d = (int)(idx % DD);
        vwt[idx] = f2bf(kv_w[(size_t)DD * HH + (size_t)d * HH + h]);
        return;
    }
    idx -= KWCNT;                   // owt[d][c] = out_w[c*D + d] (out_w is [N,H,D])
    int d = (int)(idx / NH), c = (int)(idx % NH);
    owt[idx] = f2bf(out_w[(size_t)c * DD + d]);
}

// ------------------------------------------------------------------
// 2) proj: q = x@Wq, k = x@Wk, v = x@Wv (v stored transposed [B][H][S])
//    one wave per 16x16 tile, 64 WMMA k-steps over D
// ------------------------------------------------------------------
#define QTILES 32768   // (4096/16)*(2048/16)
#define KTILES 4096    // (4096/16)*(256/16)
#define PROJ_WAVES (QTILES + 2*KTILES)

__global__ void __launch_bounds__(128) proj_kernel(
    const u16* __restrict__ xbuf, const u16* __restrict__ qwt,
    const u16* __restrict__ kwt,  const u16* __restrict__ vwt,
    u16* __restrict__ qbuf, u16* __restrict__ kbuf, u16* __restrict__ vtbuf) {
    const int lane = threadIdx.x & 31;
    int wid = ((int)blockIdx.x * (int)blockDim.x + (int)threadIdx.x) >> 5;
    if (wid >= PROJ_WAVES) return;

    int mode, mt, ct;
    const u16* wptr;
    if (wid < QTILES)            { mode = 0; mt = wid >> 7; ct = wid & 127; wptr = qwt; }
    else if (wid < QTILES+KTILES){ mode = 1; wid -= QTILES;        mt = wid >> 4; ct = wid & 15; wptr = kwt; }
    else                         { mode = 2; wid -= QTILES+KTILES; mt = wid >> 4; ct = wid & 15; wptr = vwt; }

    const int mbase = mt * 16, cbase = ct * 16;
    const int lm = lane & 15;
    const int hi = (lane >> 4) & 1;

    v8f acc = {};
    const u16* arow = xbuf + (size_t)(mbase + lm) * DD;
    const u16* brow = wptr + (size_t)(cbase + lm) * DD;
    #pragma unroll 4
    for (int ks = 0; ks < 64; ++ks) {
        const int db = ks * 32;
        v16bf a = ld_split(arow + db + hi * 8, arow + db + hi * 8 + 16);
        v16bf b = ld_split(brow + db + hi * 8, brow + db + hi * 8 + 16);
        acc = wmma_bf16(a, b, acc);
    }
    #pragma unroll
    for (int r = 0; r < 8; ++r) {
        const int m = mbase + r + hi * 8;       // token row (b*T + t)
        const int c = cbase + lm;               // output column
        const int b = m >> 11, t = m & 2047;
        const u16 val = f2bf(acc[r]);
        if (mode == 0) {
            const int n = c >> 8, h = c & 255;
            qbuf[(((size_t)(b * NQ + n)) * TT + t) * HH + h] = val;
        } else if (mode == 1) {
            kbuf[(size_t)m * HH + c] = val;
        } else {
            vtbuf[((size_t)(b * HH + c)) * TT + t] = val;
        }
    }
}

// ------------------------------------------------------------------
// 3) RoPE (+ H^-0.5 on q)
// ------------------------------------------------------------------
#define QPAIRS (BB*NQ*TT*128)
#define KPAIRS (BB*TT*128)

__global__ void __launch_bounds__(256) rope_kernel(
    const int* __restrict__ positions, u16* __restrict__ qbuf, u16* __restrict__ kbuf) {
    long long idx = (long long)blockIdx.x * blockDim.x + threadIdx.x;
    if (idx >= (long long)QPAIRS + KPAIRS) return;
    u16* base; int j, t, b; float scale;
    if (idx < QPAIRS) {
        j = (int)(idx & 127); t = (int)((idx >> 7) & 2047);
        int n = (int)((idx >> 18) & 7); b = (int)(idx >> 21);
        base = qbuf + (((size_t)(b * NQ + n)) * TT + t) * HH;
        scale = 0.0625f;                         // H^-0.5 = 1/16
    } else {
        idx -= QPAIRS;
        j = (int)(idx & 127); t = (int)((idx >> 7) & 2047); b = (int)(idx >> 18);
        base = kbuf + ((size_t)(b * TT + t)) * HH;
        scale = 1.0f;
    }
    const float pos = (float)positions[b * TT + t];
    // timescale = 10000^(j/128) ; radians = pos / timescale
    const float rad = pos * __expf(-(float)j * (9.210340371976184f / 128.0f));
    float s, c;
    __sincosf(rad, &s, &c);
    const float x1 = bf2f(base[j]);
    const float x2 = bf2f(base[j + 128]);
    base[j]       = f2bf((x1 * c - x2 * s) * scale);
    base[j + 128] = f2bf((x2 * c + x1 * s) * scale);
}

// ------------------------------------------------------------------
// 4) flash attention: one wave per (b, head, 16-row T tile)
//    logits computed TRANSPOSED (A = K rows, B = Q cols) so the f32
//    D-tile layout is exactly the bf16 A-fragment layout needed for PV.
// ------------------------------------------------------------------
#define FLASH_WAVES (BB*NQ*(TT/16))   // 2048

__global__ void __launch_bounds__(128) flash_kernel(
    const u16* __restrict__ qbuf, const u16* __restrict__ kbuf,
    const u16* __restrict__ vtbuf, u16* __restrict__ encbuf) {
    const int lane = threadIdx.x & 31;
    const int wid = ((int)blockIdx.x * (int)blockDim.x + (int)threadIdx.x) >> 5;
    if (wid >= FLASH_WAVES) return;
    const int tt = wid & 127;
    const int n  = (wid >> 7) & 7;
    const int b  = wid >> 10;
    const int tbase = tt * 16;

    const u16* qp = qbuf + ((size_t)(b * NQ + n)) * TT * HH;
    const u16* kp = kbuf + (size_t)b * TT * HH;
    const u16* vp = vtbuf + (size_t)b * HH * TT;

    const int lm = lane & 15;
    const int hi = (lane >> 4) & 1;

    v8f acc[16];
    #pragma unroll
    for (int i = 0; i < 16; ++i) acc[i] = (v8f){};

    float mrow = -3.0e38f, lrow = 0.0f;
    const int tg = tbase + lm;                    // this lane's softmax row
    const int nchunk = (tbase + 15) / 32 + 1;     // causal: s <= t only

    for (int ch = 0; ch < nchunk; ++ch) {
        const int sb = ch * 32;
        v8f c0 = {}, c1 = {};
        const u16* qrow  = qp + (size_t)(tbase + lm) * HH + hi * 8;
        const u16* krow0 = kp + (size_t)(sb + lm) * HH + hi * 8;
        const u16* krow1 = krow0 + (size_t)16 * HH;
        #pragma unroll
        for (int hs = 0; hs < 8; ++hs) {
            const int hb = hs * 32;
            v16bf qf = ld_split(qrow + hb, qrow + hb + 16);
            v16bf k0 = ld_split(krow0 + hb, krow0 + hb + 16);
            v16bf k1 = ld_split(krow1 + hb, krow1 + hb + 16);
            c0 = wmma_bf16(k0, qf, c0);   // logits^T tile, s = sb+0..15
            c1 = wmma_bf16(k1, qf, c1);   // logits^T tile, s = sb+16..31
        }
        // causal mask + online softmax stats (row = t, split over lanes l / l+16)
        float cmax = -3.0e38f;
        #pragma unroll
        for (int r = 0; r < 8; ++r) {
            const int s0 = sb + hi * 8 + r;
            if (s0 > tg)      c0[r] = -2.3819763e+38f;
            if (s0 + 16 > tg) c1[r] = -2.3819763e+38f;
            cmax = fmaxf(cmax, fmaxf(c0[r], c1[r]));
        }
        cmax = fmaxf(cmax, __shfl_xor(cmax, 16, 32));
        const float mnew = fmaxf(mrow, cmax);
        const float alpha = __expf(mrow - mnew);
        float rsum = 0.0f;
        float p0[8], p1[8];
        #pragma unroll
        for (int r = 0; r < 8; ++r) {
            p0[r] = __expf(c0[r] - mnew);
            p1[r] = __expf(c1[r] - mnew);
            rsum += p0[r] + p1[r];
        }
        rsum += __shfl_xor(rsum, 16, 32);
        lrow = alpha * lrow + rsum;
        mrow = mnew;
        // P -> bf16 A-fragment: pure in-lane pack (transposed-logits trick)
        FragU pf;
        #pragma unroll
        for (int j = 0; j < 4; ++j) pf.w[j]     = pack2(p0[2*j], p0[2*j+1]);
        #pragma unroll
        for (int j = 0; j < 4; ++j) pf.w[4 + j] = pack2(p1[2*j], p1[2*j+1]);
        // per-row rescale factors broadcast from lanes 0..15
        float fac[8];
        #pragma unroll
        for (int r = 0; r < 8; ++r) fac[r] = __shfl(alpha, r + hi * 8, 32);
        const u16* vrow = vp + (size_t)lm * TT + sb + hi * 8;
        #pragma unroll
        for (int ht = 0; ht < 16; ++ht) {
            #pragma unroll
            for (int r = 0; r < 8; ++r) acc[ht][r] *= fac[r];
            v16bf vf = ld_split(vrow + (size_t)(ht * 16) * TT, vrow + (size_t)(ht * 16) * TT + 16);
            acc[ht] = wmma_bf16(pf.v, vf, acc[ht]);
        }
    }
    float rinv[8];
    const float il = 1.0f / lrow;
    #pragma unroll
    for (int r = 0; r < 8; ++r) rinv[r] = __shfl(il, r + hi * 8, 32);
    #pragma unroll
    for (int ht = 0; ht < 16; ++ht) {
        #pragma unroll
        for (int r = 0; r < 8; ++r) {
            const int t = tbase + r + hi * 8;
            const int h = ht * 16 + lm;
            encbuf[(((size_t)(b * TT + t)) * NQ + n) * HH + h] = f2bf(acc[ht][r] * rinv[r]);
        }
    }
}

// ------------------------------------------------------------------
// 5) out projection: out[m][d] = enc[m][:] . owt[d][:]
// ------------------------------------------------------------------
#define OTILES 32768   // (4096/16)*(2048/16)

__global__ void __launch_bounds__(128) outproj_kernel(
    const u16* __restrict__ encbuf, const u16* __restrict__ owt,
    float* __restrict__ out) {
    const int lane = threadIdx.x & 31;
    const int wid = ((int)blockIdx.x * (int)blockDim.x + (int)threadIdx.x) >> 5;
    if (wid >= OTILES) return;
    const int mbase = (wid >> 7) * 16;
    const int cbase = (wid & 127) * 16;
    const int lm = lane & 15;
    const int hi = (lane >> 4) & 1;

    v8f acc = {};
    const u16* arow = encbuf + (size_t)(mbase + lm) * NH;
    const u16* brow = owt + (size_t)(cbase + lm) * NH;
    #pragma unroll 4
    for (int ks = 0; ks < 64; ++ks) {
        const int kb = ks * 32;
        v16bf a = ld_split(arow + kb + hi * 8, arow + kb + hi * 8 + 16);
        v16bf b = ld_split(brow + kb + hi * 8, brow + kb + hi * 8 + 16);
        acc = wmma_bf16(a, b, acc);
    }
    #pragma unroll
    for (int r = 0; r < 8; ++r) {
        const int m = mbase + r + hi * 8;
        const int c = cbase + lm;
        out[(size_t)m * DD + c] = acc[r];
    }
}

// ------------------------------------------------------------------
extern "C" void kernel_launch(void* const* d_in, const int* in_sizes, int n_in,
                              void* d_out, int out_size, void* d_ws, size_t ws_size,
                              hipStream_t stream) {
    const float* x     = (const float*)d_in[0];
    const int* positions = (const int*)d_in[1];
    // d_in[2] = attn_mask (causal tril) -> applied analytically
    const float* q_w   = (const float*)d_in[3];
    const float* kv_w  = (const float*)d_in[4];
    const float* out_w = (const float*)d_in[5];
    float* out = (float*)d_out;

    size_t off = 0;
    auto take = [&](size_t bytes) {
        void* p = (char*)d_ws + off;
        off += (bytes + 255) & ~(size_t)255;
        return p;
    };
    u16* xbuf   = (u16*)take((size_t)XCNT * 2);
    u16* qwt    = (u16*)take((size_t)QWCNT * 2);
    u16* kwt    = (u16*)take((size_t)KWCNT * 2);
    u16* vwt    = (u16*)take((size_t)KWCNT * 2);
    u16* owt    = (u16*)take((size_t)OWCNT * 2);
    u16* qbuf   = (u16*)take((size_t)BB * NQ * TT * HH * 2);
    u16* kbuf   = (u16*)take((size_t)BB * TT * HH * 2);
    u16* vtbuf  = (u16*)take((size_t)BB * HH * TT * 2);
    u16* encbuf = (u16*)take((size_t)BB * TT * NQ * HH * 2);

    prep_kernel<<<(PREP_TOTAL + 255) / 256, 256, 0, stream>>>(
        x, q_w, kv_w, out_w, xbuf, qwt, kwt, vwt, owt);
    proj_kernel<<<PROJ_WAVES / 4, 128, 0, stream>>>(
        xbuf, qwt, kwt, vwt, qbuf, kbuf, vtbuf);
    rope_kernel<<<(QPAIRS + KPAIRS + 255) / 256, 256, 0, stream>>>(
        positions, qbuf, kbuf);
    flash_kernel<<<FLASH_WAVES / 4, 128, 0, stream>>>(
        qbuf, kbuf, vtbuf, encbuf);
    outproj_kernel<<<OTILES / 4, 128, 0, stream>>>(encbuf, owt, out);
}
